// Involution_35304631173506
// MI455X (gfx1250) — matrix operand order repaired and checked by reading, not compiled, not run on hardware
//
#include <hip/hip_runtime.h>
#include <hip/hip_bf16.h>

typedef __attribute__((ext_vector_type(16))) __bf16 v16bf;
typedef __attribute__((ext_vector_type(8)))  float  v8f;

#define HW_   3136   // 56*56
#define C_    256
#define MID_  64
#define NOUT_ 144    // 9*16
#define W_    56

// Async global->LDS copy path (gfx1250). Guarded: falls back to plain loads
// if this toolchain doesn't expose the builtin under this name.
#if defined(__has_builtin)
#if __has_builtin(__builtin_amdgcn_global_load_async_to_lds_b32)
#define HAVE_ASYNC_LDS 1
#endif
#endif

#if HAVE_ASYNC_LDS
typedef __attribute__((address_space(1))) int* gas_i32p;
typedef __attribute__((address_space(3))) int* las_i32p;
#endif

// ---------------------------------------------------------------------------
// Kernel 1: fused  (1x1 conv + BN + ReLU) -> (1x1 conv + bias)
// One wave32 per 16-pixel M-tile. 8 waves / block. 3136 tiles total.
// GEMM1: 16x64 = sum_k(16x32 @ 32x16) over K=256   (8 K-steps, 4 N-tiles)
// GEMM2: 16x144 = sum_k(16x32 @ 32x16) over K=64   (2 K-steps, 9 N-tiles)
// ---------------------------------------------------------------------------
__global__ __launch_bounds__(256) void invo_weights_kernel(
    const float* __restrict__ x,      // (16, 256, 56, 56)
    const float* __restrict__ w1,     // (64, 256)
    const float* __restrict__ gamma,  // (64)
    const float* __restrict__ beta,   // (64)
    const float* __restrict__ mean,   // (64)
    const float* __restrict__ var,    // (64)
    const float* __restrict__ w2,     // (144, 64)
    const float* __restrict__ b2,     // (144)
    float* __restrict__ wgt)          // (16, 144, 3136) workspace
{
    // padded t-tile staging: 72*2B row pitch -> conflict-free transposed reads
    __shared__ __bf16 smem[8][16][72];

    const int lane = threadIdx.x & 31;
    const int wave = threadIdx.x >> 5;
    const int row  = lane & 15;   // A-row / B-col / C-col index
    const int half = lane >> 4;   // K-half selector

    const int tile = blockIdx.x * 8 + wave;   // 0..3135
    const int P0   = tile * 16;               // global pixel base
    const int b    = P0 / HW_;
    const int p0   = P0 - b * HW_;
    const int pix  = p0 + row;                // this lane's pixel (row M)

    const float* xb = x + (size_t)b * C_ * HW_;

    // ---------------- Stage 1: t = relu(bn(x @ w1^T)) ----------------
    v8f acc[4] = {};
    #pragma unroll
    for (int ks = 0; ks < 8; ++ks) {
        const int c0  = ks * 32;
        const int klo = c0 + half * 16;
        // A fragment: contiguous K order (lane half h -> K 16h..16h+15)
        v16bf a;
        #pragma unroll
        for (int i = 0; i < 8; ++i) {
            a[i]     = (__bf16)xb[(klo + i)      * HW_ + pix];
            a[i + 8] = (__bf16)xb[(klo + 8 + i) * HW_ + pix];
        }
        #pragma unroll
        for (int nt = 0; nt < 4; ++nt) {
            // B fragment: 32x16, lane = column o, 16 contiguous K along c
            const float* w1p = w1 + (nt * 16 + row) * C_ + c0 + half * 16;
            v16bf bfrag;
            #pragma unroll
            for (int i = 0; i < 16; ++i) bfrag[i] = (__bf16)w1p[i];
            acc[nt] = __builtin_amdgcn_wmma_f32_16x16x32_bf16(
                false, a, false, bfrag, (short)0, acc[nt], false, false);
        }
    }

    // BN (eval) + ReLU on f32 accumulators, stash t tile to LDS as bf16.
    // C layout: VGPR r, lane -> element [M = r + 8*half, N = row]
    #pragma unroll
    for (int nt = 0; nt < 4; ++nt) {
        const int o = nt * 16 + row;
        const float sc = gamma[o] * rsqrtf(var[o] + 1e-5f);
        const float bi = beta[o] - mean[o] * sc;
        #pragma unroll
        for (int r = 0; r < 8; ++r) {
            float t = acc[nt][r] * sc + bi;
            t = t > 0.f ? t : 0.f;
            smem[wave][r + half * 8][o] = (__bf16)t;
        }
    }
    __syncthreads();

    // ---------------- Stage 2: weight = t @ w2^T + b2 ----------------
    v16bf a2[2];
    #pragma unroll
    for (int ks = 0; ks < 2; ++ks) {
        const int klo = ks * 32 + half * 16;
        #pragma unroll
        for (int i = 0; i < 8; ++i) {
            a2[ks][i]     = smem[wave][row][klo + i];
            a2[ks][i + 8] = smem[wave][row][klo + 8 + i];
        }
    }

    #pragma unroll
    for (int nt = 0; nt < 9; ++nt) {
        const int o = nt * 16 + row;
        v8f acc2 = {};
        #pragma unroll
        for (int ks = 0; ks < 2; ++ks) {
            const float* w2p = w2 + o * MID_ + ks * 32 + half * 16;
            v16bf bfrag;
            #pragma unroll
            for (int i = 0; i < 16; ++i) bfrag[i] = (__bf16)w2p[i];
            acc2 = __builtin_amdgcn_wmma_f32_16x16x32_bf16(
                false, a2[ks], false, bfrag, (short)0, acc2, false, false);
        }
        const float bias = b2[o];
        float* wp = wgt + ((size_t)b * NOUT_ + o) * HW_ + p0;
        #pragma unroll
        for (int r = 0; r < 8; ++r)
            wp[r + half * 8] = acc2[r] + bias;
    }
}

// ---------------------------------------------------------------------------
// Kernel 2: LDS-tiled involution apply.
// Block = (b, g, 8x28 pixel tile). 56 = 8*7 = 28*2 -> exact tiling.
// Stage x[16ch][10r][30w] halo tile into LDS (async-to-LDS when available),
// per-pixel weights live in registers and are reused across the 16 channels.
// ---------------------------------------------------------------------------
#define TH_ 8
#define TW_ 28
#define HR_ 10   // TH_ + 2
#define HC_ 30   // TW_ + 2
#define NTHREADS2 (TH_ * TW_)   // 224 = 7 waves
#define NELEM (16 * HR_ * HC_)  // 4800 floats staged per block

__global__ __launch_bounds__(NTHREADS2) void invo_apply_kernel(
    const float* __restrict__ x,
    const float* __restrict__ wgt,
    float* __restrict__ out)
{
    __shared__ float xs[16][HR_][HC_];   // 19.2 KB

    const int tid = threadIdx.x;
    const int tileid = blockIdx.x;       // 0..13
    const int g = blockIdx.y;            // 0..15
    const int b = blockIdx.z;            // 0..15
    const int ty = tileid >> 1;          // 0..6
    const int tx = tileid & 1;           // 0..1
    const int h0 = ty * TH_;
    const int w0 = tx * TW_;

    const float* xg = x + ((size_t)b * C_ + g * 16) * HW_;

    // ---- stage halo tile: in-bounds cells copied (async), OOB cells zeroed
    //      (disjoint addresses -> no DS-vs-async ordering hazard) ----
    #pragma unroll
    for (int k = 0; k < (NELEM + NTHREADS2 - 1) / NTHREADS2; ++k) {
        const int e = tid + k * NTHREADS2;
        if (e < NELEM) {
            const int c  = e / (HR_ * HC_);
            const int rr = (e / HC_) % HR_;
            const int cc = e % HC_;
            const int gh = h0 - 1 + rr;
            const int gw = w0 - 1 + cc;
            float* lp = &xs[c][rr][cc];
            if (gh >= 0 && gh < W_ && gw >= 0 && gw < W_) {
                const float* gp = xg + c * HW_ + gh * W_ + gw;
#if HAVE_ASYNC_LDS
                __builtin_amdgcn_global_load_async_to_lds_b32(
                    (gas_i32p)(void*)gp, (las_i32p)(void*)lp, 0, 0);
#else
                *lp = *gp;
#endif
            } else {
                *lp = 0.f;
            }
        }
    }
#if HAVE_ASYNC_LDS
#if __has_builtin(__builtin_amdgcn_s_wait_asynccnt)
    __builtin_amdgcn_s_wait_asynccnt(0);
#else
    asm volatile("s_wait_asynccnt 0" ::: "memory");
#endif
#endif
    __syncthreads();

    // ---- compute: one thread per pixel, 16 channels each ----
    const int py = tid / TW_;            // 0..7
    const int px = tid - py * TW_;       // 0..27
    const int h = h0 + py;
    const int w = w0 + px;
    const int p = h * W_ + w;

    // 9 per-pixel kernel weights, coalesced loads, reused across 16 channels
    const float* wp = wgt + ((size_t)b * NOUT_ + g * 9) * HW_ + p;
    float wv[9];
    #pragma unroll
    for (int kk = 0; kk < 9; ++kk) wv[kk] = wp[kk * HW_];

    float* outp = out + ((size_t)b * C_ + g * 16) * HW_ + p;
    #pragma unroll
    for (int c = 0; c < 16; ++c) {
        float sum = 0.f;
        #pragma unroll
        for (int dy = 0; dy < 3; ++dy) {
            const float* xr = &xs[c][py + dy][px];
            sum = fmaf(wv[dy * 3 + 0], xr[0], sum);
            sum = fmaf(wv[dy * 3 + 1], xr[1], sum);
            sum = fmaf(wv[dy * 3 + 2], xr[2], sum);
        }
        outp[(size_t)c * HW_] = sum;
    }
}

extern "C" void kernel_launch(void* const* d_in, const int* in_sizes, int n_in,
                              void* d_out, int out_size, void* d_ws, size_t ws_size,
                              hipStream_t stream) {
    const float* x     = (const float*)d_in[0];
    const float* w1    = (const float*)d_in[1];
    const float* gamma = (const float*)d_in[2];
    const float* beta  = (const float*)d_in[3];
    const float* mean  = (const float*)d_in[4];
    const float* var   = (const float*)d_in[5];
    const float* w2    = (const float*)d_in[6];
    const float* b2    = (const float*)d_in[7];
    float* wgt = (float*)d_ws;                 // needs 16*144*3136*4 = 27.6 MB
    float* out = (float*)d_out;

    // 3136 M-tiles, 8 waves (tiles) per block
    invo_weights_kernel<<<392, 256, 0, stream>>>(x, w1, gamma, beta, mean, var,
                                                 w2, b2, wgt);
    // (tile, group, batch) grid; 224 threads = one per pixel of the 8x28 tile
    invo_apply_kernel<<<dim3(14, 16, 16), NTHREADS2, 0, stream>>>(x, wgt, out);
}
